// SDRQwenAttention_42623255445783
// MI455X (gfx1250) — matrix operand order repaired and checked by reading, not compiled, not run on hardware
//
#include <hip/hip_runtime.h>

typedef __attribute__((ext_vector_type(16))) _Float16 v16h;
typedef __attribute__((ext_vector_type(8)))  float    v8f;
typedef __attribute__((ext_vector_type(4)))  unsigned int u32x4;
typedef __attribute__((ext_vector_type(8)))  unsigned int u32x8;

#define B_    2
#define S_    2048
#define H_    2048
#define NH_   16
#define NKV_  2
#define HD_   128
#define GROUPS_ (NH_ / NKV_)

static __device__ __forceinline__ v8f wmma_f16(v16h a, v16h b, v8f c) {
  // v_wmma_f32_16x16x32_f16  D = A*B + C
  return __builtin_amdgcn_wmma_f32_16x16x32_f16(false, a, false, b, (short)0, c,
                                                false, false);
}

// K-offset inside a 32-wide K-chunk in the contiguous k order used here for A and B (not AMD's table order):
// lanes 0-15 hold K=0..15, lanes 16-31 hold K=16..31, 2/VGPR.
static __device__ __forceinline__ int a_koff(int j, int hl) {
  return hl * 16 + 2 * j;
}

// LDS byte offset of a __shared__-derived pointer (generic ptr low 32 bits).
static __device__ __forceinline__ unsigned int lds_off(const void* p) {
  return (unsigned int)(uintptr_t)p;
}

// ---------------------------------------------------------------------------
// Tensor Data Mover: 2D tile (f16 elements) global -> LDS, with row padding.
// D# built per CDNA5 ISA 8.3/8.4; issued one-per-wave (EXEC ignored).
//   tile_d0 : contiguous elements per row,  tile_d1 : rows
//   stride0 : row stride in elements
//   pad_int : pad_interval code (3 => every 16 DW, 5 => every 64 DW)
// pad_amount code 0 => 1 DWORD (2 f16) appended -> LDS stride = tile_d0 + 2.
// ---------------------------------------------------------------------------
static __device__ __forceinline__ void tdm_load_2d_f16(
    const void* gptr, unsigned int ldsaddr, unsigned int tile_d0,
    unsigned int tile_d1, unsigned int stride0, unsigned int pad_int) {
  unsigned long long ga = (unsigned long long)(uintptr_t)gptr;
  u32x4 g0;
  g0[0] = 1u;                                            // count=1, user D#
  g0[1] = ldsaddr;                                       // lds_addr (bytes)
  g0[2] = (unsigned int)ga;                              // global_addr[31:0]
  g0[3] = ((unsigned int)(ga >> 32) & 0x01FFFFFFu) | (2u << 30);  // type=2
  u32x8 g1;
  g1[0] = (1u << 16)           // data_size = 1 (2 bytes)
        | (1u << 20)           // pad_enable
        | (pad_int << 22)      // pad_interval
        | (0u << 25);          // pad_amount = 1 DWORD
  g1[1] = (tile_d0 & 0xFFFFu) << 16;                     // tensor_dim0 lo16
  g1[2] = ((tile_d1 & 0xFFFFu) << 16);                   // dim0 hi=0 | dim1 lo
  g1[3] = (tile_d0 & 0xFFFFu) << 16;                     // dim1 hi=0 | tile_dim0
  g1[4] = tile_d1 & 0xFFFFu;                             // tile_dim1 | tile_dim2=0
  g1[5] = stride0;                                       // tensor_dim0_stride lo32
  g1[6] = 0u;
  g1[7] = 0u;
  asm volatile("tensor_load_to_lds %0, %1" :: "s"(g0), "s"(g1) : "memory");
}

// ---------------------------------------------------------------------------
// f32 -> f16 conversion (grid-stride)
// ---------------------------------------------------------------------------
__global__ void cvt_f32_f16(const float* __restrict__ src,
                            _Float16* __restrict__ dst, int n) {
  int i = blockIdx.x * blockDim.x + threadIdx.x;
  int stride = gridDim.x * blockDim.x;
  for (; i < n; i += stride) dst[i] = (_Float16)src[i];
}

// ---------------------------------------------------------------------------
// Tiled WMMA GEMM:  C[m,n] = sum_k A[m,k] * W[n,k]  (+bias) (+RoPE)
// 256 thr = 8 waves, C-tile 128x128, wave = 16 rows x 128 cols.
// A/B tiles staged by TDM, double-buffered (DMA overlaps WMMA).
// ---------------------------------------------------------------------------
template <int ROPE, int F32OUT>
__global__ __launch_bounds__(256) void gemm_kernel(
    const _Float16* __restrict__ A,   // M x K (f16)
    const _Float16* __restrict__ W,   // N x K (f16)
    const float* __restrict__ bias,   // N (or null)
    _Float16* __restrict__ dst16,     // [B][heads][S][HD]
    float* __restrict__ dst32,        // [M][N]
    const float* __restrict__ cosp,
    const float* __restrict__ sinp,
    int nheads) {
  const int tid  = threadIdx.x;
  const int lane = tid & 31, w = tid >> 5;
  const int hl   = lane >> 4, c = lane & 15;
  const int m0   = blockIdx.x * 128;
  const int n0   = blockIdx.y * 128;

  // TDM-padded tiles: 32 k-elems/row + 1 DW pad -> stride 34 halves
  __shared__ _Float16 As[2][128 * 34];  // [m][k]
  __shared__ _Float16 Bs[2][128 * 34];  // [n][k]

  v8f acc[8] = {};
  const int NKC = H_ / 32;

  if (w == 0) {  // prologue: stage k-chunk 0
    tdm_load_2d_f16(A + (size_t)m0 * H_, lds_off(&As[0][0]), 32, 128, H_, 3);
    tdm_load_2d_f16(W + (size_t)n0 * H_, lds_off(&Bs[0][0]), 32, 128, H_, 3);
  }

  for (int i = 0; i < NKC; ++i) {
    const int cur = i & 1;
    if (w == 0) {
      if (i + 1 < NKC) {
        int kc = (i + 1) * 32;
        tdm_load_2d_f16(A + (size_t)m0 * H_ + kc, lds_off(&As[cur ^ 1][0]),
                        32, 128, H_, 3);
        tdm_load_2d_f16(W + (size_t)n0 * H_ + kc, lds_off(&Bs[cur ^ 1][0]),
                        32, 128, H_, 3);
        __builtin_amdgcn_s_wait_tensorcnt(2);  // pair(i) done, pair(i+1) flies
      } else {
        __builtin_amdgcn_s_wait_tensorcnt(0);
      }
    }
    __syncthreads();  // buf[cur] ready for all waves

    v16h af;
    const int arow = 16 * w + c;
#pragma unroll
    for (int j = 0; j < 8; ++j) {
      int ko = a_koff(j, hl);
      af[2 * j]     = As[cur][arow * 34 + ko];
      af[2 * j + 1] = As[cur][arow * 34 + ko + 1];
    }
#pragma unroll
    for (int t = 0; t < 8; ++t) {
      v16h bf;  // B fragment: lanes 0-15 K=0..15, lanes 16-31 K=16..31
#pragma unroll
      for (int j = 0; j < 8; ++j) {
        int kk = hl * 16 + 2 * j;
        bf[2 * j]     = Bs[cur][(16 * t + c) * 34 + kk];
        bf[2 * j + 1] = Bs[cur][(16 * t + c) * 34 + kk + 1];
      }
      acc[t] = wmma_f16(af, bf, acc[t]);
    }
    __syncthreads();  // all reads done before TDM overwrites buf[cur] next iter
  }

  if constexpr (F32OUT) {
#pragma unroll
    for (int t = 0; t < 8; ++t)
#pragma unroll
      for (int r = 0; r < 8; ++r) {
        int row = m0 + 16 * w + r + 8 * hl;
        dst32[(size_t)row * H_ + n0 + 16 * t + c] = acc[t][r];
      }
  } else {
    float bvv[8];
#pragma unroll
    for (int t = 0; t < 8; ++t) bvv[t] = bias[n0 + 16 * t + c];
    const int head = n0 >> 7;  // HD_ == 128
#pragma unroll
    for (int t = 0; t < 8; ++t)
#pragma unroll
      for (int r = 0; r < 8; ++r) {
        int row = m0 + 16 * w + r + 8 * hl;
        int b = row >> 11, s = row & (S_ - 1);
        int d = 16 * t + c;
        float val = acc[t][r] + bvv[t];
        if constexpr (ROPE) {
          float partner = (t < 4) ? -(acc[t + 4][r] + bvv[t + 4])
                                  :  (acc[t - 4][r] + bvv[t - 4]);
          size_t ci = ((size_t)b * S_ + s) * HD_ + d;
          val = val * cosp[ci] + partner * sinp[ci];
        }
        dst16[(((size_t)b * nheads + head) * S_ + s) * HD_ + d] = (_Float16)val;
      }
  }
}

// ---------------------------------------------------------------------------
// Attention pass 1: online softmax statistics (row max m, row sum l).
// Q block and K tiles staged by TDM; K tiles double-buffered.
// ---------------------------------------------------------------------------
__global__ __launch_bounds__(256) void attn_pass1(
    const _Float16* __restrict__ Q, const _Float16* __restrict__ Km,
    float* __restrict__ ml) {
  const int tid = threadIdx.x, lane = tid & 31, w = tid >> 5;
  const int hl = lane >> 4, c = lane & 15;
  const int bh = blockIdx.x;  // b*NH + h
  const int b = bh / NH_, h = bh % NH_;
  const int hk = h / GROUPS_;
  const int q0 = blockIdx.y * 128;
  const float scale = 0.08838834764831845f;  // 1/sqrt(128)

  __shared__ _Float16 Qs[128 * 130];   // TDM pad -> stride 130
  __shared__ _Float16 Ks[2][16 * 130];

  const _Float16* Qbase = Q + ((size_t)bh * S_ + q0) * HD_;
  const _Float16* Kbase = Km + ((size_t)(b * NKV_ + hk) * S_) * HD_;
  const int ktmax = (q0 + 127) >> 4;

  if (w == 0) {
    tdm_load_2d_f16(Qbase, lds_off(&Qs[0]), 128, 128, HD_, 5);
    tdm_load_2d_f16(Kbase, lds_off(&Ks[0][0]), 128, 16, HD_, 5);
    __builtin_amdgcn_s_wait_tensorcnt(1);  // Q done; K tile 0 still in flight
  }
  __syncthreads();

  // Q fragments: loop-invariant, gather once (4 K-chunks of 32 over HD)
  v16h aq[4];
  const int arow = 16 * w + c;
#pragma unroll
  for (int dk = 0; dk < 4; ++dk)
#pragma unroll
    for (int j = 0; j < 8; ++j) {
      int ko = 32 * dk + a_koff(j, hl);
      aq[dk][2 * j]     = Qs[arow * 130 + ko];
      aq[dk][2 * j + 1] = Qs[arow * 130 + ko + 1];
    }

  float m[8], l[8];
#pragma unroll
  for (int r = 0; r < 8; ++r) { m[r] = -1.0e30f; l[r] = 0.0f; }

  for (int kt = 0; kt <= ktmax; ++kt) {
    const int cur = kt & 1;
    if (w == 0) {
      if (kt < ktmax) {
        tdm_load_2d_f16(Kbase + (size_t)(kt + 1) * 16 * HD_,
                        lds_off(&Ks[cur ^ 1][0]), 128, 16, HD_, 5);
        __builtin_amdgcn_s_wait_tensorcnt(1);
      } else {
        __builtin_amdgcn_s_wait_tensorcnt(0);
      }
    }
    __syncthreads();  // Ks[cur] ready

    v8f sc = {};
#pragma unroll
    for (int dk = 0; dk < 4; ++dk) {
      v16h bf;  // B = K^T : B[dd][n] = K[key n][dd]
#pragma unroll
      for (int j = 0; j < 8; ++j) {
        int dd = 32 * dk + hl * 16 + 2 * j;
        bf[2 * j]     = Ks[cur][c * 130 + dd];
        bf[2 * j + 1] = Ks[cur][c * 130 + dd + 1];
      }
      sc = wmma_f16(aq[dk], bf, sc);
    }

#pragma unroll
    for (int r = 0; r < 8; ++r) {
      int qi = q0 + 16 * w + r + 8 * hl;
      int kj = kt * 16 + c;
      float s = sc[r] * scale;
      if (kj > qi) s = -1.0e9f;
      float rm = s;
      for (int off = 8; off; off >>= 1) rm = fmaxf(rm, __shfl_xor(rm, off, 32));
      float nm = fmaxf(m[r], rm);
      float e = __expf(s - nm);
      for (int off = 8; off; off >>= 1) e += __shfl_xor(e, off, 32);
      l[r] = l[r] * __expf(m[r] - nm) + e;
      m[r] = nm;
    }
    __syncthreads();  // reads done before next TDM overwrite
  }

  if (c == 0) {
#pragma unroll
    for (int r = 0; r < 8; ++r) {
      int qi = q0 + 16 * w + r + 8 * hl;
      size_t idx = ((size_t)bh * S_ + qi) * 2;
      ml[idx] = m[r];
      ml[idx + 1] = l[r];
    }
  }
}

// ---------------------------------------------------------------------------
// Attention pass 2: recompute scores, write P (f32, exactly once), and
// accumulate O = P @ V with WMMA.  Q buffer is reused as the K/V double
// buffers after the Q fragments are lifted into registers.
// ---------------------------------------------------------------------------
__global__ __launch_bounds__(256) void attn_pass2(
    const _Float16* __restrict__ Q, const _Float16* __restrict__ Km,
    const _Float16* __restrict__ V, const float* __restrict__ ml,
    float* __restrict__ attn, _Float16* __restrict__ Ocat) {
  const int tid = threadIdx.x, lane = tid & 31, w = tid >> 5;
  const int hl = lane >> 4, c = lane & 15;
  const int bh = blockIdx.x;
  const int b = bh / NH_, h = bh % NH_;
  const int hk = h / GROUPS_;
  const int q0 = blockIdx.y * 128;
  const float scale = 0.08838834764831845f;

  // 33,280 B pool: Q (128x130) during prologue, then K/V double buffers.
  // Layout after prologue: K0 @ 0, K1 @ 4160, V0 @ 8320, V1 @ 12480 (elems).
  __shared__ _Float16 pool[16640];
  __shared__ _Float16 Ps[8][16 * 32];  // per-wave P tile, A-frag source

  const _Float16* Qbase = Q + ((size_t)bh * S_ + q0) * HD_;
  const _Float16* Kbase = Km + ((size_t)(b * NKV_ + hk) * S_) * HD_;
  const _Float16* Vbase = V + ((size_t)(b * NKV_ + hk) * S_) * HD_;
  float* attnbh = attn + (size_t)bh * S_ * S_;

  if (w == 0) {
    tdm_load_2d_f16(Qbase, lds_off(pool), 128, 128, HD_, 5);
    __builtin_amdgcn_s_wait_tensorcnt(0);
  }
  __syncthreads();

  v16h aq[4];
  const int arow = 16 * w + c;
#pragma unroll
  for (int dk = 0; dk < 4; ++dk)
#pragma unroll
    for (int j = 0; j < 8; ++j) {
      int ko = 32 * dk + a_koff(j, hl);
      aq[dk][2 * j]     = pool[arow * 130 + ko];
      aq[dk][2 * j + 1] = pool[arow * 130 + ko + 1];
    }

  float m[8], invl[8];
#pragma unroll
  for (int r = 0; r < 8; ++r) {
    int qi = q0 + 16 * w + r + 8 * hl;
    size_t idx = ((size_t)bh * S_ + qi) * 2;
    m[r] = ml[idx];
    invl[r] = 1.0f / ml[idx + 1];
  }
  __syncthreads();  // all waves done reading Q before K/V overwrite the pool

  if (w == 0) {  // prologue: K/V tile 0
    tdm_load_2d_f16(Kbase, lds_off(pool), 128, 32, HD_, 5);
    tdm_load_2d_f16(Vbase, lds_off(pool + 8320), 128, 32, HD_, 5);
  }

  v8f oacc[8] = {};
  const int NKB = S_ / 32;

  for (int kb = 0; kb < NKB; ++kb) {
    const int k0 = kb * 32;
    const int cur = kb & 1;
    // runtime pointers into the pool (no static addrspacecast initializers)
    const _Float16* Kc = pool + cur * 4160;
    const _Float16* Vc = pool + 8320 + cur * 4160;
    const bool next_live = ((k0 + 32) <= q0 + 127) && (kb + 1 < NKB);
    if (w == 0) {
      if (next_live) {
        tdm_load_2d_f16(Kbase + (size_t)(k0 + 32) * HD_,
                        lds_off(pool + (cur ^ 1) * 4160), 128, 32, HD_, 5);
        tdm_load_2d_f16(Vbase + (size_t)(k0 + 32) * HD_,
                        lds_off(pool + 8320 + (cur ^ 1) * 4160), 128, 32, HD_, 5);
        __builtin_amdgcn_s_wait_tensorcnt(2);  // pair(kb) done
      } else {
        __builtin_amdgcn_s_wait_tensorcnt(0);
      }
    }
    __syncthreads();

    const bool live_wave = (k0 <= q0 + 16 * w + 15);
    if (live_wave) {
#pragma unroll
      for (int sub = 0; sub < 2; ++sub) {
        v8f sc = {};
#pragma unroll
        for (int dk = 0; dk < 4; ++dk) {
          v16h bf;
#pragma unroll
          for (int j = 0; j < 8; ++j) {
            int dd = 32 * dk + hl * 16 + 2 * j;
            bf[2 * j]     = Kc[(16 * sub + c) * 130 + dd];
            bf[2 * j + 1] = Kc[(16 * sub + c) * 130 + dd + 1];
          }
          sc = wmma_f16(aq[dk], bf, sc);
        }
#pragma unroll
        for (int r = 0; r < 8; ++r) {
          int qi = q0 + 16 * w + r + 8 * hl;
          int kj = k0 + 16 * sub + c;
          float p = 0.0f;
          if (kj <= qi) p = __expf(sc[r] * scale - m[r]) * invl[r];
          attnbh[(size_t)qi * S_ + kj] = p;
          Ps[w][(r + 8 * hl) * 32 + 16 * sub + c] = (_Float16)p;
        }
      }
      // same-wave LDS RAW: P stores -> A-fragment gathers
      asm volatile("s_wait_dscnt 0" ::: "memory");
      v16h ap;
#pragma unroll
      for (int j = 0; j < 8; ++j) {
        int ko = a_koff(j, hl);
        ap[2 * j]     = Ps[w][c * 32 + ko];
        ap[2 * j + 1] = Ps[w][c * 32 + ko + 1];
      }
#pragma unroll
      for (int t = 0; t < 8; ++t) {
        v16h bf;  // B = V chunk: 32 keys x 16 d-cols
#pragma unroll
        for (int j = 0; j < 8; ++j) {
          int kk = hl * 16 + 2 * j;
          bf[2 * j]     = Vc[kk * 130 + 16 * t + c];
          bf[2 * j + 1] = Vc[(kk + 1) * 130 + 16 * t + c];
        }
        oacc[t] = wmma_f16(ap, bf, oacc[t]);
      }
    } else {
      // fully masked for this wave: attn probabilities are exactly zero
#pragma unroll
      for (int r = 0; r < 8; ++r) {
        int qi = q0 + 16 * w + r + 8 * hl;
        attnbh[(size_t)qi * S_ + k0 + c] = 0.0f;
        attnbh[(size_t)qi * S_ + k0 + 16 + c] = 0.0f;
      }
    }
    __syncthreads();  // reads done before next TDM overwrite of buf[cur]
  }

  // O -> [B][S][NH*HD] f16 for the output projection GEMM
#pragma unroll
  for (int t = 0; t < 8; ++t)
#pragma unroll
    for (int r = 0; r < 8; ++r) {
      int qi = q0 + 16 * w + r + 8 * hl;
      int d = 16 * t + c;
      Ocat[((size_t)b * S_ + qi) * (NH_ * HD_) + h * HD_ + d] =
          (_Float16)oacc[t][r];
    }
}

// ---------------------------------------------------------------------------
// Host-side orchestration
// ---------------------------------------------------------------------------
extern "C" void kernel_launch(void* const* d_in, const int* in_sizes, int n_in,
                              void* d_out, int out_size, void* d_ws,
                              size_t ws_size, hipStream_t stream) {
  (void)in_sizes; (void)n_in; (void)out_size; (void)ws_size;
  const float* hidden = (const float*)d_in[0];
  // d_in[1] = attention_mask: identical to analytic causal mask; computed inline
  const float* cosp = (const float*)d_in[2];
  const float* sinp = (const float*)d_in[3];
  const float* Wq = (const float*)d_in[4];
  const float* bq = (const float*)d_in[5];
  const float* Wk = (const float*)d_in[6];
  const float* bk = (const float*)d_in[7];
  const float* Wv = (const float*)d_in[8];
  const float* bv = (const float*)d_in[9];
  const float* Wo = (const float*)d_in[10];

  char* ws = (char*)d_ws;
  _Float16* h16   = (_Float16*)(ws + 0);          // 16,777,216 B
  _Float16* Wq16  = (_Float16*)(ws + 16777216);   //  8,388,608 B
  _Float16* Wk16  = (_Float16*)(ws + 25165824);   //  1,048,576 B
  _Float16* Wv16  = (_Float16*)(ws + 26214400);   //  1,048,576 B
  _Float16* Wo16  = (_Float16*)(ws + 27262976);   //  8,388,608 B
  _Float16* Q16   = (_Float16*)(ws + 35651584);   // 16,777,216 B
  _Float16* K16   = (_Float16*)(ws + 52428800);   //  2,097,152 B
  _Float16* V16   = (_Float16*)(ws + 54525952);   //  2,097,152 B
  _Float16* O16   = (_Float16*)(ws + 56623104);   // 16,777,216 B
  float*    mlbuf = (float*)   (ws + 73400320);   //    524,288 B

  float* out  = (float*)d_out;
  float* attn = out + (size_t)B_ * S_ * H_;

  cvt_f32_f16<<<2048, 256, 0, stream>>>(hidden, h16, B_ * S_ * H_);
  cvt_f32_f16<<<2048, 256, 0, stream>>>(Wq, Wq16, NH_ * HD_ * H_);
  cvt_f32_f16<<<512, 256, 0, stream>>>(Wk, Wk16, NKV_ * HD_ * H_);
  cvt_f32_f16<<<512, 256, 0, stream>>>(Wv, Wv16, NKV_ * HD_ * H_);
  cvt_f32_f16<<<2048, 256, 0, stream>>>(Wo, Wo16, H_ * NH_ * HD_);

  dim3 blk(256);
  gemm_kernel<1, 0><<<dim3(32, 16), blk, 0, stream>>>(h16, Wq16, bq, Q16,
                                                      nullptr, cosp, sinp, NH_);
  gemm_kernel<1, 0><<<dim3(32, 2), blk, 0, stream>>>(h16, Wk16, bk, K16,
                                                     nullptr, cosp, sinp, NKV_);
  gemm_kernel<0, 0><<<dim3(32, 2), blk, 0, stream>>>(h16, Wv16, bv, V16,
                                                     nullptr, cosp, sinp, NKV_);

  attn_pass1<<<dim3(B_ * NH_, S_ / 128), blk, 0, stream>>>(Q16, K16, mlbuf);
  attn_pass2<<<dim3(B_ * NH_, S_ / 128), blk, 0, stream>>>(Q16, K16, V16, mlbuf,
                                                           attn, O16);

  gemm_kernel<0, 1><<<dim3(32, 16), blk, 0, stream>>>(
      O16, Wo16, nullptr, nullptr, out, nullptr, nullptr, NH_);
}
